// Block_43379169690252
// MI455X (gfx1250) — matrix operand order repaired and checked by reading, not compiled, not run on hardware
//
#include <hip/hip_runtime.h>
#include <hip/hip_bf16.h>

#define HEADS   16
#define DMODEL  1024
#define DEXP    2048
#define DHEAD   128
#define SEQ     2048
#define DOUT    (4*DEXP + HEADS)   // 8208

typedef __attribute__((ext_vector_type(16))) _Float16 v16h;
typedef __attribute__((ext_vector_type(8)))  _Float16 v8h;
typedef __attribute__((ext_vector_type(8)))  float    v8f;
typedef __attribute__((ext_vector_type(4)))  unsigned int u32x4;
typedef __attribute__((ext_vector_type(8)))  int      i32x8;
typedef __attribute__((ext_vector_type(4)))  int      i32x4;

#if defined(__has_builtin)
#if __has_builtin(__builtin_amdgcn_tensor_load_to_lds) && \
    __has_builtin(__builtin_amdgcn_s_wait_tensorcnt)
#define USE_TDM 1
#endif
#endif

// ---------------------------------------------------------------------------
// WMMA helpers (CDNA5 V_WMMA_F32_16X16X32_F16, wave32)
// ---------------------------------------------------------------------------
__device__ __forceinline__ v16h mk16(v8h lo, v8h hi) {
    return __builtin_shufflevector(lo, hi, 0,1,2,3,4,5,6,7,8,9,10,11,12,13,14,15);
}

// A-matrix 16x32 f16 fragment from a [16][ld] (row-major, M x K) tile.
// Per ISA 7.12.2: lane<16 holds row m=lane, K={0..7}u{16..23};
//                 lane>=16 holds row m=lane-16, K={8..15}u{24..31}.
__device__ __forceinline__ v16h ldA(const _Float16* tile, int ld, int lane) {
    int m = lane & 15, hf = lane >> 4;
    const _Float16* p0 = tile + m * ld + hf * 8;
    return mk16(*(const v8h*)p0, *(const v8h*)(p0 + 16));
}

// B-matrix 32x16 f16 fragment from an [N][K] (row-major, 16 x 32) tile.
// lane<16: column n=lane, K={0..7}u{16..23}; lane>=16: K={8..15}u{24..31}.
__device__ __forceinline__ v16h ldB(const _Float16* tile, int ld, int lane) {
    int n = lane & 15, hf = lane >> 4;
    const _Float16* p0 = tile + n * ld + hf * 8;
    return mk16(*(const v8h*)p0, *(const v8h*)(p0 + 16));
}

__device__ __forceinline__ v8f wmma_f16(v16h a, v16h b, v8f c) {
    return __builtin_amdgcn_wmma_f32_16x16x32_f16(
        /*neg_a=*/false, a, /*neg_b=*/false, b,
        /*c_mod=*/(short)0, c, /*reuse_a=*/false, /*reuse_b=*/false);
}

__device__ __forceinline__ float sigmoidf(float x) { return 1.0f / (1.0f + __expf(-x)); }

// ---------------------------------------------------------------------------
// LayerNorm (fp32 in), f16 or f32 out. One block per row of 1024.
// ---------------------------------------------------------------------------
__device__ __forceinline__ float block_reduce(float v, float* red, int t) {
    red[t] = v; __syncthreads();
    for (int o = 128; o > 0; o >>= 1) { if (t < o) red[t] += red[t + o]; __syncthreads(); }
    float r = red[0]; __syncthreads();
    return r;
}

__global__ __launch_bounds__(256) void ln_in_kernel(const float* __restrict__ x,
                                                    const float* __restrict__ w,
                                                    const float* __restrict__ b,
                                                    _Float16* __restrict__ out) {
    const int row = blockIdx.x, t = threadIdx.x;
    const float* xr = x + (size_t)row * DMODEL;
    __shared__ float red[256];
    float s = 0.f, s2 = 0.f;
    for (int c = t; c < DMODEL; c += 256) { float v = xr[c]; s += v; s2 += v * v; }
    float S  = block_reduce(s,  red, t);
    float S2 = block_reduce(s2, red, t);
    float mu  = S * (1.0f / DMODEL);
    float var = S2 * (1.0f / DMODEL) - mu * mu;
    float rs  = rsqrtf(var + 1e-5f);
    for (int c = t; c < DMODEL; c += 256)
        out[(size_t)row * DMODEL + c] = (_Float16)((xr[c] - mu) * rs * w[c] + b[c]);
}

__global__ __launch_bounds__(256) void ln_out_kernel(const float* __restrict__ x,
                                                     const float* __restrict__ w,
                                                     const float* __restrict__ b,
                                                     float* __restrict__ out) {
    const int row = blockIdx.x, t = threadIdx.x;
    const float* xr = x + (size_t)row * DMODEL;
    __shared__ float red[256];
    float s = 0.f, s2 = 0.f;
    for (int c = t; c < DMODEL; c += 256) { float v = xr[c]; s += v; s2 += v * v; }
    float S  = block_reduce(s,  red, t);
    float S2 = block_reduce(s2, red, t);
    float mu  = S * (1.0f / DMODEL);
    float var = S2 * (1.0f / DMODEL) - mu * mu;
    float rs  = rsqrtf(var + 1e-5f);
    for (int c = t; c < DMODEL; c += 256)
        out[(size_t)row * DMODEL + c] = (xr[c] - mu) * rs * w[c] + b[c];
}

// ---------------------------------------------------------------------------
// fp32 -> f16 convert
// ---------------------------------------------------------------------------
__global__ __launch_bounds__(256) void cvt_f16_kernel(const float* __restrict__ src,
                                                      _Float16* __restrict__ dst, int n) {
    int i = blockIdx.x * 256 + threadIdx.x;
    if (i < n) dst[i] = (_Float16)src[i];
}

// ---------------------------------------------------------------------------
// Projection GEMM: proj = xn(2048x1024,f16) @ W_in^T(8208x1024,f16) + b_in
// Block tile 128(M) x 64(N), K-step 32. 8 waves, each 16(M) x 64(N).
// ---------------------------------------------------------------------------
__global__ __launch_bounds__(256) void gemm_in_kernel(const _Float16* __restrict__ xnh,
                                                      const _Float16* __restrict__ winh,
                                                      const float* __restrict__ b_in,
                                                      const float* __restrict__ log_scale,
                                                      _Float16* __restrict__ qh,
                                                      _Float16* __restrict__ kraw,
                                                      _Float16* __restrict__ vh,
                                                      float* __restrict__ pbuf,
                                                      float* __restrict__ ysig) {
    __shared__ _Float16 sA[128 * 32];
    __shared__ _Float16 sB[64 * 32];
    const int tid = threadIdx.x, lane = tid & 31, wv = tid >> 5;
    const int i0 = blockIdx.y * 128, j0 = blockIdx.x * 64;

    v8f c[4];
    #pragma unroll
    for (int sn = 0; sn < 4; ++sn)
        #pragma unroll
        for (int r = 0; r < 8; ++r) c[sn][r] = 0.f;

    for (int kk = 0; kk < DMODEL; kk += 32) {
        __syncthreads();
        {   // A tile: 128x32 = 4096 halfs, 16 per thread
            int idx = tid * 16, row = idx >> 5, col = idx & 31;
            const _Float16* src = xnh + (size_t)(i0 + row) * DMODEL + kk + col;
            *(v8h*)(sA + idx)     = *(const v8h*)src;
            *(v8h*)(sA + idx + 8) = *(const v8h*)(src + 8);
            __builtin_prefetch((const void*)(src + 32), 0, 1);
        }
        {   // B tile: 64x32 = 2048 halfs, 8 per thread; guard N edge (8208)
            int idx = tid * 8, row = idx >> 5, col = idx & 31;
            int j = j0 + row;
            v8h val;
            if (j < DOUT) {
                val = *(const v8h*)(winh + (size_t)j * DMODEL + kk + col);
                __builtin_prefetch((const void*)(winh + (size_t)j * DMODEL + kk + col + 32), 0, 1);
            } else {
                #pragma unroll
                for (int e = 0; e < 8; ++e) val[e] = (_Float16)0.f;
            }
            *(v8h*)(sB + idx) = val;
        }
        __syncthreads();
        v16h af = ldA(sA + (wv * 16) * 32, 32, lane);
        v16h bf[4];
        #pragma unroll
        for (int sn = 0; sn < 4; ++sn) bf[sn] = ldB(sB + (sn * 16) * 32, 32, lane);
        #pragma unroll
        for (int sn = 0; sn < 4; ++sn) c[sn] = wmma_f16(af, bf[sn], c[sn]);
    }

    // Epilogue routing. C layout: n = lane&15, row m = r + 8*(lane>>4).
    const int ln = lane & 15, hf = lane >> 4;
    const float inv_sqrt_d = 0.08838834764831845f;  // 1/sqrt(128)
    #pragma unroll
    for (int sn = 0; sn < 4; ++sn) {
        int j = j0 + sn * 16 + ln;
        if (j >= DOUT) continue;
        float bias = b_in[j];
        #pragma unroll
        for (int r = 0; r < 8; ++r) {
            int i = i0 + wv * 16 + r + 8 * hf;
            float val = c[sn][r] + bias;
            if (j < DEXP) {                        // q, fold 1/(s^2 * sqrt(d))
                int hh = j >> 7, d = j & 127;
                float s = __expf(log_scale[hh]);
                qh[((size_t)hh * SEQ + i) * DHEAD + d] =
                    (_Float16)(val * (1.0f / (s * s)) * inv_sqrt_d);
            } else if (j < 2 * DEXP) {             // k (pre-smear)
                int jj = j - DEXP; int hh = jj >> 7, d = jj & 127;
                kraw[((size_t)hh * SEQ + i) * DHEAD + d] = (_Float16)val;
            } else if (j < 3 * DEXP) {             // v
                int jj = j - 2 * DEXP; int hh = jj >> 7, d = jj & 127;
                vh[((size_t)hh * SEQ + i) * DHEAD + d] = (_Float16)val;
            } else if (j < 4 * DEXP) {             // p gate (raw, f32)
                pbuf[(size_t)i * DEXP + (j - 3 * DEXP)] = val;
            } else {                               // y -> sigmoid(y)
                ysig[(size_t)i * HEADS + (j - 4 * DEXP)] = sigmoidf(val);
            }
        }
    }
}

// ---------------------------------------------------------------------------
// Key smear: k_t <- (1-s)*k_t + s*k_{t-1}, s = sigmoid(smear_factor[h])
// ---------------------------------------------------------------------------
__global__ __launch_bounds__(256) void smear_kernel(const _Float16* __restrict__ kraw,
                                                    const float* __restrict__ smear_factor,
                                                    _Float16* __restrict__ ksm) {
    size_t idx = (size_t)blockIdx.x * 256 + threadIdx.x;   // 16*2048*128 total
    int h = (int)(idx >> 18);
    int rem = (int)(idx & 262143);
    int i = rem >> 7;
    float s = sigmoidf(smear_factor[h]);
    float cur  = (float)kraw[idx];
    float prev = (i > 0) ? (float)kraw[idx - DHEAD] : 0.f;
    ksm[idx] = (_Float16)((1.f - s) * cur + s * prev);
}

// ---------------------------------------------------------------------------
// pos[h][i] = cumsum_i sigmoid(y)[i][h]. One wave32 per head, shfl scan.
// ---------------------------------------------------------------------------
__global__ __launch_bounds__(512) void pos_kernel(const float* __restrict__ ysig,
                                                  float* __restrict__ pos) {
    const int lane = threadIdx.x & 31, h = threadIdx.x >> 5;   // 16 waves
    float carry = 0.f;
    for (int c = 0; c < SEQ / 32; ++c) {
        int i = c * 32 + lane;
        float v = ysig[(size_t)i * HEADS + h];
        #pragma unroll
        for (int off = 1; off < 32; off <<= 1) {
            float t = __shfl_up(v, off, 32);
            if (lane >= off) v += t;
        }
        pos[(size_t)h * SEQ + i] = carry + v;
        carry += __shfl(v, 31, 32);
    }
}

// ---------------------------------------------------------------------------
// Flash attention. Block = 1 head x 128 queries, 8 waves (16 q-rows each).
// K tile (32x128 f16, no transpose) is DMA'd by the Tensor Data Mover when
// available (wave 0 issues; TENSORcnt + barrier publishes); V is staged
// transposed ([d][key]) for the O-phase B fragments.
// ---------------------------------------------------------------------------
__global__ __launch_bounds__(256) void attn_kernel(const _Float16* __restrict__ qh,
                                                   const _Float16* __restrict__ kh,
                                                   const _Float16* __restrict__ vh,
                                                   const float* __restrict__ pos,
                                                   float* __restrict__ obuf) {
    const int h  = blockIdx.y;
    const int i0 = blockIdx.x * 128;
    const int tid = threadIdx.x, lane = tid & 31, wv = tid >> 5;
    const int ln = lane & 15, hf = lane >> 4;

    __shared__ _Float16 sK [32 * DHEAD];    // [key][d]  -> B for S = Q K^T
    __shared__ _Float16 sVT[DHEAD * 32];    // [d][key]  -> B for O += P V
    __shared__ _Float16 sP [8][16 * 32];    // per-wave P tile re-fragment
    __shared__ float    sPos[SEQ];

    for (int t = tid; t < SEQ; t += 256) sPos[t] = pos[(size_t)h * SEQ + t];

    // Q strip for this wave: 16 rows x 128, four A fragments (K-step 32)
    v16h qf[4];
    {
        const _Float16* qbase = qh + ((size_t)h * SEQ + i0 + wv * 16) * DHEAD;
        #pragma unroll
        for (int t = 0; t < 4; ++t) qf[t] = ldA(qbase + t * 32, DHEAD, lane);
    }

    v8f oacc[8];
    float rmax[8], rsum[8], pos_i[8];
    #pragma unroll
    for (int st = 0; st < 8; ++st)
        #pragma unroll
        for (int r = 0; r < 8; ++r) oacc[st][r] = 0.f;

    __syncthreads();   // sPos visible
    #pragma unroll
    for (int r = 0; r < 8; ++r) {
        int i = i0 + wv * 16 + r + 8 * hf;
        pos_i[r] = sPos[i];
        rmax[r] = -INFINITY;
        rsum[r] = 0.f;
    }

    const int jmax = i0 + 127;
    for (int j0 = 0; j0 <= jmax; j0 += 32) {
        __syncthreads();
#ifdef USE_TDM
        if (wv == 0) {
            // Tensor DMA: 32x128 f16 tile of K[h] starting at row j0 -> sK.
            // D# per CDNA5 ISA ch.8: group0 = {count, lds_addr, global_addr,
            // type=2}; group1 = {data_size=2B, tensor_dim0=128,
            // tensor_dim1=SEQ, tile 128x32, dim0_stride=128}.
            unsigned lds_off = (unsigned)(size_t)(void*)sK;
            unsigned long long ga =
                (unsigned long long)(size_t)(kh + ((size_t)h * SEQ + j0) * DHEAD);
            u32x4 g0;
            g0[0] = 1u;                                          // count=1
            g0[1] = lds_off;                                     // lds_addr
            g0[2] = (unsigned)(ga & 0xFFFFFFFFu);                // global lo
            g0[3] = (unsigned)((ga >> 32) & 0x01FFFFFFu) | (2u << 30); // type=2
            i32x8 g1;
            g1[0] = (int)(1u << 16);          // data_size = 2 bytes
            g1[1] = (int)(128u << 16);        // tensor_dim0 = 128
            g1[2] = (int)((unsigned)SEQ << 16); // tensor_dim1 = 2048
            g1[3] = (int)(128u << 16);        // tile_dim0 = 128
            g1[4] = 32;                       // tile_dim1 = 32
            g1[5] = 128;                      // tensor_dim0_stride = 128
            g1[6] = 0; g1[7] = 0;
            i32x4 gz = {0, 0, 0, 0};
#if defined(__clang_major__) && (__clang_major__ >= 23)
            i32x8 gz8 = {0, 0, 0, 0, 0, 0, 0, 0};
            __builtin_amdgcn_tensor_load_to_lds(g0, g1, gz, gz, gz8, 0);
#else
            __builtin_amdgcn_tensor_load_to_lds(g0, g1, gz, gz, 0);
#endif
            __builtin_amdgcn_s_wait_tensorcnt(0);
        }
#else
        {   // K tile 32x128, 16 halfs/thread
            int idx = tid * 16, key = idx >> 7, d = idx & 127;
            const _Float16* src = kh + ((size_t)h * SEQ + j0 + key) * DHEAD + d;
            *(v8h*)(sK + idx)     = *(const v8h*)src;
            *(v8h*)(sK + idx + 8) = *(const v8h*)(src + 8);
            __builtin_prefetch((const void*)(src + 32 * DHEAD), 0, 1);
        }
#endif
        {   // V tile transposed into [d][key]
            int idx = tid * 16, key = idx >> 7, d = idx & 127;
            const _Float16* src = vh + ((size_t)h * SEQ + j0 + key) * DHEAD + d;
            v8h a = *(const v8h*)src, b = *(const v8h*)(src + 8);
            __builtin_prefetch((const void*)(src + 32 * DHEAD), 0, 1);
            #pragma unroll
            for (int e = 0; e < 8; ++e) {
                sVT[(d + e)     * 32 + key] = a[e];
                sVT[(d + 8 + e) * 32 + key] = b[e];
            }
        }
        __syncthreads();

        // Scores: two 16-key column tiles, contraction over d=128
        v8f s0, s1;
        #pragma unroll
        for (int r = 0; r < 8; ++r) { s0[r] = 0.f; s1[r] = 0.f; }
        #pragma unroll
        for (int t = 0; t < 4; ++t) {
            v16h b0 = ldB(sK + t * 32,              DHEAD, lane);
            v16h b1 = ldB(sK + 16 * DHEAD + t * 32, DHEAD, lane);
            s0 = wmma_f16(qf[t], b0, s0);
            s1 = wmma_f16(qf[t], b1, s1);
        }

        float pj0 = sPos[j0 + ln], pj1 = sPos[j0 + 16 + ln];
        #pragma unroll
        for (int r = 0; r < 8; ++r) {
            int i = i0 + wv * 16 + r + 8 * hf;
            float e0 = s0[r] + ((j0 + ln      > i) ? -1e10f : 0.f) - (pos_i[r] - pj0);
            float e1 = s1[r] + ((j0 + 16 + ln > i) ? -1e10f : 0.f) - (pos_i[r] - pj1);
            float m = fmaxf(e0, e1);
            #pragma unroll
            for (int off = 1; off < 16; off <<= 1) m = fmaxf(m, __shfl_xor(m, off, 16));
            float nm = fmaxf(rmax[r], m);
            float scale = __expf(rmax[r] - nm);
            float p0 = __expf(e0 - nm), p1 = __expf(e1 - nm);
            float ps = p0 + p1;
            #pragma unroll
            for (int off = 1; off < 16; off <<= 1) ps += __shfl_xor(ps, off, 16);
            rsum[r] = rsum[r] * scale + ps;
            rmax[r] = nm;
            #pragma unroll
            for (int st = 0; st < 8; ++st) oacc[st][r] *= scale;
            sP[wv][(r + 8 * hf) * 32 + ln]      = (_Float16)p0;
            sP[wv][(r + 8 * hf) * 32 + ln + 16] = (_Float16)p1;
        }
        // Re-fragment P (16x32) as A; O += P * V across 8 d-subtiles.
        v16h pa = ldA(sP[wv], 32, lane);
        #pragma unroll
        for (int st = 0; st < 8; ++st) {
            v16h bv = ldB(sVT + (st * 16) * 32, 32, lane);
            oacc[st] = wmma_f16(pa, bv, oacc[st]);
        }
    }

    // Epilogue: normalize and write o[i][h*128+d] (fp32)
    #pragma unroll
    for (int st = 0; st < 8; ++st)
        #pragma unroll
        for (int r = 0; r < 8; ++r) {
            int i = i0 + wv * 16 + r + 8 * hf;
            int d = st * 16 + ln;
            obuf[(size_t)i * DEXP + h * DHEAD + d] = oacc[st][r] / rsum[r];
        }
}

// ---------------------------------------------------------------------------
// Gate: og = f16( silu(p) * o )
// ---------------------------------------------------------------------------
__global__ __launch_bounds__(256) void gate_kernel(const float* __restrict__ pbuf,
                                                   const float* __restrict__ obuf,
                                                   _Float16* __restrict__ og) {
    size_t idx = (size_t)blockIdx.x * 256 + threadIdx.x;   // 2048*2048
    float p = pbuf[idx];
    og[idx] = (_Float16)(p * sigmoidf(p) * obuf[idx]);
}

// ---------------------------------------------------------------------------
// Output GEMM: out(2048x1024,f32) = og(2048x2048,f16) @ W_out^T(1024x2048,f16)
// ---------------------------------------------------------------------------
__global__ __launch_bounds__(256) void gemm_out_kernel(const _Float16* __restrict__ og,
                                                       const _Float16* __restrict__ wouth,
                                                       float* __restrict__ out) {
    __shared__ _Float16 sA[128 * 32];
    __shared__ _Float16 sB[64 * 32];
    const int tid = threadIdx.x, lane = tid & 31, wv = tid >> 5;
    const int i0 = blockIdx.y * 128, j0 = blockIdx.x * 64;

    v8f c[4];
    #pragma unroll
    for (int sn = 0; sn < 4; ++sn)
        #pragma unroll
        for (int r = 0; r < 8; ++r) c[sn][r] = 0.f;

    for (int kk = 0; kk < DEXP; kk += 32) {
        __syncthreads();
        {
            int idx = tid * 16, row = idx >> 5, col = idx & 31;
            const _Float16* src = og + (size_t)(i0 + row) * DEXP + kk + col;
            *(v8h*)(sA + idx)     = *(const v8h*)src;
            *(v8h*)(sA + idx + 8) = *(const v8h*)(src + 8);
            __builtin_prefetch((const void*)(src + 32), 0, 1);
        }
        {
            int idx = tid * 8, row = idx >> 5, col = idx & 31;
            const _Float16* src = wouth + (size_t)(j0 + row) * DEXP + kk + col;
            *(v8h*)(sB + idx) = *(const v8h*)src;
            __builtin_prefetch((const void*)(src + 32), 0, 1);
        }
        __syncthreads();
        v16h af = ldA(sA + (wv * 16) * 32, 32, lane);
        v16h bf[4];
        #pragma unroll
        for (int sn = 0; sn < 4; ++sn) bf[sn] = ldB(sB + (sn * 16) * 32, 32, lane);
        #pragma unroll
        for (int sn = 0; sn < 4; ++sn) c[sn] = wmma_f16(af, bf[sn], c[sn]);
    }

    const int ln = lane & 15, hf = lane >> 4;
    #pragma unroll
    for (int sn = 0; sn < 4; ++sn) {
        int j = j0 + sn * 16 + ln;
        #pragma unroll
        for (int r = 0; r < 8; ++r) {
            int i = i0 + wv * 16 + r + 8 * hf;
            out[(size_t)i * DMODEL + j] = c[sn][r];
        }
    }
}

// ---------------------------------------------------------------------------
// Workspace layout (bytes)
// ---------------------------------------------------------------------------
static constexpr size_t OFF_XN   = 0;                                         // 2048x1024 f16
static constexpr size_t OFF_WIN  = OFF_XN   + (size_t)SEQ * DMODEL * 2;       // 8208x1024 f16
static constexpr size_t OFF_WOUT = OFF_WIN  + (size_t)DOUT * DMODEL * 2;      // 1024x2048 f16
static constexpr size_t OFF_Q    = OFF_WOUT + (size_t)DMODEL * DEXP * 2;      // 16x2048x128 f16
static constexpr size_t OFF_KRAW = OFF_Q    + (size_t)HEADS * SEQ * DHEAD * 2;
static constexpr size_t OFF_KSM  = OFF_KRAW + (size_t)HEADS * SEQ * DHEAD * 2;
static constexpr size_t OFF_V    = OFF_KSM  + (size_t)HEADS * SEQ * DHEAD * 2;
static constexpr size_t OFF_P    = OFF_V    + (size_t)HEADS * SEQ * DHEAD * 2; // 2048x2048 f32
static constexpr size_t OFF_YS   = OFF_P    + (size_t)SEQ * DEXP * 4;          // 2048x16 f32
static constexpr size_t OFF_POS  = OFF_YS   + (size_t)SEQ * HEADS * 4;         // 16x2048 f32
static constexpr size_t OFF_O    = OFF_POS  + (size_t)HEADS * SEQ * 4;         // 2048x2048 f32
static constexpr size_t OFF_OG   = OFF_O    + (size_t)SEQ * DEXP * 4;          // 2048x2048 f16
static constexpr size_t OFF_OUT  = OFF_OG   + (size_t)SEQ * DEXP * 2;          // 2048x1024 f32

extern "C" void kernel_launch(void* const* d_in, const int* in_sizes, int n_in,
                              void* d_out, int out_size, void* d_ws, size_t ws_size,
                              hipStream_t stream) {
    (void)in_sizes; (void)n_in; (void)out_size; (void)ws_size;

    const float* x            = (const float*)d_in[0];
    const float* W_in         = (const float*)d_in[1];
    const float* b_in         = (const float*)d_in[2];
    const float* in_ln_w      = (const float*)d_in[3];
    const float* in_ln_b      = (const float*)d_in[4];
    const float* W_out        = (const float*)d_in[5];
    const float* out_ln_w     = (const float*)d_in[6];
    const float* out_ln_b     = (const float*)d_in[7];
    const float* smear_factor = (const float*)d_in[8];
    const float* log_scale    = (const float*)d_in[9];
    float* out = (float*)d_out;

    char* ws = (char*)d_ws;
    _Float16* xn_h   = (_Float16*)(ws + OFF_XN);
    _Float16* win_h  = (_Float16*)(ws + OFF_WIN);
    _Float16* wout_h = (_Float16*)(ws + OFF_WOUT);
    _Float16* q_h    = (_Float16*)(ws + OFF_Q);
    _Float16* kraw_h = (_Float16*)(ws + OFF_KRAW);
    _Float16* ksm_h  = (_Float16*)(ws + OFF_KSM);
    _Float16* v_h    = (_Float16*)(ws + OFF_V);
    float*    p_f    = (float*)(ws + OFF_P);
    float*    ysig   = (float*)(ws + OFF_YS);
    float*    pos    = (float*)(ws + OFF_POS);
    float*    o_f    = (float*)(ws + OFF_O);
    _Float16* og_h   = (_Float16*)(ws + OFF_OG);
    float*    out_f  = (float*)(ws + OFF_OUT);

    // 1. input LN -> f16
    ln_in_kernel<<<SEQ, 256, 0, stream>>>(x, in_ln_w, in_ln_b, xn_h);

    // 2. weight conversion
    {
        int n1 = DOUT * DMODEL;
        cvt_f16_kernel<<<(n1 + 255) / 256, 256, 0, stream>>>(W_in, win_h, n1);
        int n2 = DMODEL * DEXP;
        cvt_f16_kernel<<<(n2 + 255) / 256, 256, 0, stream>>>(W_out, wout_h, n2);
    }

    // 3. fused projection GEMM (WMMA)
    gemm_in_kernel<<<dim3((DOUT + 63) / 64, SEQ / 128), 256, 0, stream>>>(
        xn_h, win_h, b_in, log_scale, q_h, kraw_h, v_h, p_f, ysig);

    // 4. key smear + position cumsum
    smear_kernel<<<(HEADS * SEQ * DHEAD) / 256, 256, 0, stream>>>(kraw_h, smear_factor, ksm_h);
    pos_kernel<<<1, 512, 0, stream>>>(ysig, pos);

    // 5. flash attention (WMMA + TDM K-tile streaming)
    attn_kernel<<<dim3(SEQ / 128, HEADS), 256, 0, stream>>>(q_h, ksm_h, v_h, pos, o_f);

    // 6. SiLU gate
    gate_kernel<<<(SEQ * DEXP) / 256, 256, 0, stream>>>(p_f, o_f, og_h);

    // 7. output GEMM (WMMA)
    gemm_out_kernel<<<dim3(DMODEL / 64, SEQ / 128), 256, 0, stream>>>(og_h, wout_h, out_f);

    // 8. output LN -> d_out (fp32)
    ln_out_kernel<<<SEQ, 256, 0, stream>>>(out_f, out_ln_w, out_ln_b, out);
}
